// StateWeightNet_12438225289584
// MI455X (gfx1250) — matrix operand order repaired and checked by reading, not compiled, not run on hardware
//
#include <hip/hip_runtime.h>
#include <hip/hip_bf16.h>

typedef __attribute__((ext_vector_type(16))) _Float16 v16h;
typedef __attribute__((ext_vector_type(8)))  _Float16 v8h;
typedef __attribute__((ext_vector_type(8)))  float    v8f;

#define NNODES 70000
#define NEDGES 1120000
#define DFEAT  73
#define WPB    4        // waves per block in GEMM kernel (wave32)

// ---------- ordered-float <-> uint encoding for atomicMax on floats ----------
__device__ __forceinline__ unsigned fenc(float f) {
    unsigned u = __float_as_uint(f);
    return (u & 0x80000000u) ? ~u : (u | 0x80000000u);
}
__device__ __forceinline__ float fdec(unsigned u) {
    return (u & 0x80000000u) ? __uint_as_float(u & 0x7FFFFFFFu)
                             : __uint_as_float(~u);
}

// =====================================================================
// Templated WMMA GEMM: C[:, coff:coff+NOUT] = act(A[nrows,K] @ W[NOUT,K]^T + b)
// One wave = one 16-row tile. A tile and the whole (padded) W staged in
// LDS as f16; fragments built with contiguous ds_load_b128 vector loads.
// =====================================================================
template<int K, int NOUT>
__global__ void gemm_bias_act(const float* __restrict__ A, int lda,
                              const float* __restrict__ W,
                              const float* __restrict__ bias,
                              float* __restrict__ C, int ldc, int coff,
                              int nrows, int act)
{
    constexpr int KP  = (K + 31) & ~31;   // padded K (mult of 32)
    constexpr int NKC = KP / 32;          // k-chunks
    constexpr int NT  = (NOUT + 15) / 16; // out-tiles
    constexpr int WR  = NT * 16;          // padded out rows

    __shared__ __attribute__((aligned(32))) _Float16 sA[WPB][16 * KP];
    __shared__ __attribute__((aligned(32))) _Float16 sW[WR * KP];

    const int lane = threadIdx.x & 31;
    const int wid  = threadIdx.x >> 5;
    const int tile = blockIdx.x * WPB + wid;
    const int row0 = tile * 16;
    const int hl   = lane >> 4;           // half-wave (0/1)
    const int l15  = lane & 15;
    const bool active = (row0 < nrows);   // nrows is a multiple of 16

    __builtin_prefetch(W, 0, 1);          // global_prefetch_b8

    // ---- stage W (zero-padded) into LDS, whole block, branch-free ----
    for (int idx = threadIdx.x; idx < WR * KP; idx += blockDim.x) {
        int r  = idx / KP, kk = idx - r * KP;
        int rc = r  < NOUT ? r  : NOUT - 1;
        int kc = kk < K    ? kk : K    - 1;
        float v = W[(size_t)rc * K + kc];
        sW[idx] = (_Float16)((r < NOUT && kk < K) ? v : 0.0f);
    }
    // ---- stage A tile (f32 -> f16) into LDS, per wave, branch-free ----
    if (active) {
        const float* Arow = A + (size_t)row0 * lda;
        for (int idx = lane; idx < 16 * KP; idx += 32) {
            int r  = idx / KP, kk = idx - r * KP;
            int kc = kk < K ? kk : K - 1;
            float v = Arow[(size_t)r * lda + kc];
            sA[wid][idx] = (_Float16)(kk < K ? v : 0.0f);
        }
    }
    __syncthreads();
    if (!active) return;

    // ---- hoist A fragments (16-bit A 16x32 layout: two 8-half runs) ----
    v16h av[NKC];
#pragma unroll
    for (int kc = 0; kc < NKC; ++kc) {
        const _Float16* p = &sA[wid][l15 * KP + kc * 32 + 8 * hl];
        v8h a0 = *(const v8h*)p;            // K = kb+8*hl   .. +8
        v8h a1 = *(const v8h*)(p + 16);     // K = kb+8*hl+16.. +8
        av[kc] = __builtin_shufflevector(a0, a1,
                     0,1,2,3,4,5,6,7,8,9,10,11,12,13,14,15);
    }

#pragma unroll
    for (int ot = 0; ot < NT; ++ot) {
        v8f c = {};
        const int ocol = ot * 16 + l15;
#pragma unroll
        for (int kc = 0; kc < NKC; ++kc) {
            // B fragment: lane = N col, two 8-half K runs (8*hl, 8*hl+16)
            const _Float16* q = &sW[ocol * KP + kc * 32 + 8 * hl];
            v16h b = __builtin_shufflevector(*(const v8h*)q, *(const v8h*)(q + 16),
                     0,1,2,3,4,5,6,7,8,9,10,11,12,13,14,15);
            c = __builtin_amdgcn_wmma_f32_16x16x32_f16(
                    false, av[kc], false, b, (short)0, c, false, false);
        }
        if (ocol < NOUT) {
            float bv = (bias != nullptr) ? bias[ocol] : 0.0f;
#pragma unroll
            for (int r = 0; r < 8; ++r) {
                int row = row0 + r + 8 * hl;   // C layout: VGPR r -> M=r(+8)
                float v = c[r] + bv;
                if (act) v = v > 0.0f ? v : 0.0f;
                C[(size_t)row * ldc + coff + ocol] = v;
            }
        }
    }
}

// =====================================================================
// Fill / init kernels
// =====================================================================
__global__ void k_fill_zero(float* __restrict__ p, int n) {
    int i = blockIdx.x * blockDim.x + threadIdx.x;
    if (i < n) p[i] = 0.0f;
}
__global__ void k_init_ms(unsigned* __restrict__ menc, float* __restrict__ sbuf, int n) {
    int i = blockIdx.x * blockDim.x + threadIdx.x;
    if (i < n) { menc[i] = fenc(-__builtin_inff()); sbuf[i] = 0.0f; }
}

// =====================================================================
// Edge kernels (wave-per-edge: coalesced 73-float row reads, shfl reduce)
// =====================================================================
__global__ void k_edge_logits(const float* __restrict__ xl,
                              const float* __restrict__ xr,
                              const float* __restrict__ att,
                              const int* __restrict__ src,
                              const int* __restrict__ dst,
                              float* __restrict__ ebuf,
                              unsigned* __restrict__ menc, int E)
{
    int edge = (blockIdx.x * blockDim.x + threadIdx.x) >> 5;
    int lane = threadIdx.x & 31;
    if (edge >= E) return;
    int si = src[edge], di = dst[edge];
    const float* pl = xl + (size_t)si * DFEAT;
    const float* pr = xr + (size_t)di * DFEAT;
    float acc = 0.0f;
    for (int k = lane; k < DFEAT; k += 32) {
        float h = pl[k] + pr[k];
        h = h > 0.0f ? h : 0.2f * h;        // leaky_relu(0.2)
        acc += h * att[k];
    }
#pragma unroll
    for (int off = 16; off > 0; off >>= 1) acc += __shfl_xor(acc, off, 32);
    if (lane == 0) {
        ebuf[edge] = acc;
        atomicMax(menc + di, fenc(acc));
    }
}

__global__ void k_decode_m(float* __restrict__ mbuf, int n) {
    int i = blockIdx.x * blockDim.x + threadIdx.x;
    if (i >= n) return;
    unsigned* menc = (unsigned*)mbuf;
    float m = fdec(menc[i]);
    unsigned e = (__float_as_uint(m) >> 23) & 0xFFu;
    if (e == 0xFFu) m = 0.0f;               // guard empty segments (inf/NaN -> 0)
    mbuf[i] = m;
}

__global__ void k_exp_sum(float* __restrict__ ebuf,
                          const float* __restrict__ mbuf,
                          const int* __restrict__ dst,
                          float* __restrict__ sbuf, int E)
{
    int i = blockIdx.x * blockDim.x + threadIdx.x;
    if (i >= E) return;
    int di = dst[i];
    float a = __expf(ebuf[i] - mbuf[di]);
    ebuf[i] = a;
    atomicAdd(sbuf + di, a);
}

__global__ void k_scatter(const float* __restrict__ ebuf,
                          const float* __restrict__ sbuf,
                          const float* __restrict__ xl,
                          const int* __restrict__ src,
                          const int* __restrict__ dst,
                          float* __restrict__ g, int coff, int E)
{
    int edge = (blockIdx.x * blockDim.x + threadIdx.x) >> 5;
    int lane = threadIdx.x & 31;
    if (edge >= E) return;
    int si = src[edge], di = dst[edge];
    float alpha = ebuf[edge] / (sbuf[di] + 1e-16f);
    const float* pl = xl + (size_t)si * DFEAT;
    float* pg = g + (size_t)di * 146 + coff;
    for (int k = lane; k < DFEAT; k += 32)
        atomicAdd(pg + k, alpha * pl[k]);
}

__global__ void k_finalize(float* __restrict__ g,
                           const float* __restrict__ bias,
                           const float* __restrict__ x, int coff, int total)
{
    int i = blockIdx.x * blockDim.x + threadIdx.x;
    if (i >= total) return;
    int n = i / DFEAT, k = i - n * DFEAT;
    float v = g[(size_t)n * 146 + coff + k] + bias[k] + x[(size_t)n * DFEAT + k];
    g[(size_t)n * 146 + coff + k] = v > 0.0f ? v : 0.0f;  // relu(gat + x)
}

// =====================================================================
// MLP head last layer: out[n] = h3[n,:32] . W4 + b4
// =====================================================================
__global__ void k_head(const float* __restrict__ h3,
                       const float* __restrict__ W4,
                       const float* __restrict__ b4,
                       float* __restrict__ out, int n)
{
    int i = blockIdx.x * blockDim.x + threadIdx.x;
    if (i >= n) return;
    float acc = b4[0];
    const float* p = h3 + (size_t)i * 32;
#pragma unroll
    for (int k = 0; k < 32; ++k) acc += p[k] * W4[k];
    out[i] = acc;
}

// =====================================================================
// Launch
// =====================================================================
extern "C" void kernel_launch(void* const* d_in, const int* in_sizes, int n_in,
                              void* d_out, int out_size, void* d_ws, size_t ws_size,
                              hipStream_t stream) {
    const int N = NNODES, E = NEDGES;
    const float* x_f  = (const float*)d_in[0];
    const int*   ei_f = (const int*)  d_in[1];
    const float* x_b  = (const float*)d_in[2];
    const int*   ei_b = (const int*)  d_in[3];
    const float* Wl[2] = { (const float*)d_in[4],  (const float*)d_in[10] };
    const float* bl[2] = { (const float*)d_in[5],  (const float*)d_in[11] };
    const float* Wr[2] = { (const float*)d_in[6],  (const float*)d_in[12] };
    const float* br[2] = { (const float*)d_in[7],  (const float*)d_in[13] };
    const float* at[2] = { (const float*)d_in[8],  (const float*)d_in[14] };
    const float* bi[2] = { (const float*)d_in[9],  (const float*)d_in[15] };
    const float* W1 = (const float*)d_in[16]; const float* b1 = (const float*)d_in[17];
    const float* W2 = (const float*)d_in[18]; const float* b2 = (const float*)d_in[19];
    const float* W3 = (const float*)d_in[20]; const float* b3 = (const float*)d_in[21];
    const float* W4 = (const float*)d_in[22]; const float* b4 = (const float*)d_in[23];
    const float* xs[2]  = { x_f, x_b };
    const int*   eis[2] = { ei_f, ei_b };

    // ---- workspace arena (floats), with aliasing for MLP intermediates ----
    float* ws   = (float*)d_ws;
    float* g    = ws;                       // [N,146]
    float* xl   = g  + (size_t)N * 146;     // [N,73]
    float* xr   = xl + (size_t)N * 73;      // [N,73]
    float* eb   = xr + (size_t)N * 73;      // [E]
    float* mbuf = eb + (size_t)E;           // [N]  (unsigned during max pass)
    float* sbuf = mbuf + (size_t)N;         // [N]
    float* h1   = xl;                       // [N,128] alias (xl/xr dead after GAT)
    float* h2   = g;                        // [N,64]  alias (g dead after L1)
    float* h3   = g + (size_t)N * 64;       // [N,32]  alias

    const int TPB = 256;
    const int gemm_blocks = (((N + 15) / 16) + WPB - 1) / WPB;

    k_fill_zero<<<(N * 146 + TPB - 1) / TPB, TPB, 0, stream>>>(g, N * 146);

    for (int d = 0; d < 2; ++d) {
        const float* x = xs[d];
        const int* src = eis[d];
        const int* dst = eis[d] + E;
        const int coff = d * DFEAT;
        // node transforms via WMMA
        gemm_bias_act<DFEAT, DFEAT><<<gemm_blocks, 32 * WPB, 0, stream>>>(
            x, DFEAT, Wl[d], bl[d], xl, DFEAT, 0, N, 0);
        gemm_bias_act<DFEAT, DFEAT><<<gemm_blocks, 32 * WPB, 0, stream>>>(
            x, DFEAT, Wr[d], br[d], xr, DFEAT, 0, N, 0);
        // segment softmax + weighted scatter
        k_init_ms<<<(N + TPB - 1) / TPB, TPB, 0, stream>>>((unsigned*)mbuf, sbuf, N);
        k_edge_logits<<<((size_t)E * 32 + TPB - 1) / TPB, TPB, 0, stream>>>(
            xl, xr, at[d], src, dst, eb, (unsigned*)mbuf, E);
        k_decode_m<<<(N + TPB - 1) / TPB, TPB, 0, stream>>>(mbuf, N);
        k_exp_sum<<<(E + TPB - 1) / TPB, TPB, 0, stream>>>(eb, mbuf, dst, sbuf, E);
        k_scatter<<<((size_t)E * 32 + TPB - 1) / TPB, TPB, 0, stream>>>(
            eb, sbuf, xl, src, dst, g, coff, E);
        k_finalize<<<(N * DFEAT + TPB - 1) / TPB, TPB, 0, stream>>>(
            g, bi[d], x, coff, N * DFEAT);
    }

    // MLP head (WMMA for the three big layers)
    gemm_bias_act<146, 128><<<gemm_blocks, 32 * WPB, 0, stream>>>(g,  146, W1, b1, h1, 128, 0, N, 1);
    gemm_bias_act<128,  64><<<gemm_blocks, 32 * WPB, 0, stream>>>(h1, 128, W2, b2, h2,  64, 0, N, 1);
    gemm_bias_act< 64,  32><<<gemm_blocks, 32 * WPB, 0, stream>>>(h2,  64, W3, b3, h3,  32, 0, N, 1);
    k_head<<<(N + TPB - 1) / TPB, TPB, 0, stream>>>(h3, W4, b4, (float*)d_out, N);
}
